// MemoryAsContextTransformer_16338055594768
// MI455X (gfx1250) — matrix operand order repaired and checked by reading, not compiled, not run on hardware
//
#include <hip/hip_runtime.h>

// ---------------------------------------------------------------------------
// MemoryAsContextTransformer for MI455X (gfx1250), wave32 + WMMA bf16 +
// async global->LDS staging (ASYNCcnt).
// Shapes fixed by reference: B=2, S=4096, D=1024, HEADS=16, DIM_HEAD=64,
// SEG=512, P=16  =>  pad=0, nw=8, tokens=8192, inner=1024.
// ---------------------------------------------------------------------------

typedef __attribute__((ext_vector_type(16))) __bf16        v16bf;
typedef __attribute__((ext_vector_type(8)))  float         v8f;
typedef __attribute__((ext_vector_type(8)))  unsigned int  v8u;

__device__ __forceinline__ unsigned short f2bf(float f) {
    unsigned u = __builtin_bit_cast(unsigned, f);
    unsigned r = u + 0x7FFFu + ((u >> 16) & 1u);   // round-to-nearest-even
    return (unsigned short)(r >> 16);
}

__device__ __forceinline__ v8f wmma_bf16(v8u a, v8u b, v8f c) {
    return __builtin_amdgcn_wmma_f32_16x16x32_bf16(
        false, __builtin_bit_cast(v16bf, a),
        false, __builtin_bit_cast(v16bf, b),
        (short)0, c, false, false);
}

// LDS byte address of a __shared__ object: LDS aperture keeps the offset in
// addr[31:0] (ISA 10.2), so truncating the generic pointer is the offset.
__device__ __forceinline__ unsigned lds_addr(const void* p) {
    return (unsigned)(unsigned long long)p;
}

// CDNA5 async copy: 16 bytes per lane, global -> LDS, tracked by ASYNCcnt.
__device__ __forceinline__ void async_copy_b128(unsigned ldsAddr, const void* gaddr) {
    asm volatile("global_load_async_to_lds_b128 %0, %1, off"
                 :: "v"(ldsAddr), "v"(gaddr) : "memory");
}
__device__ __forceinline__ void wait_async0() {
    asm volatile("s_wait_asynccnt 0" ::: "memory");
}

// A-fragment (16x32 bf16, M x K) from row-major ushort src, row stride ld.
// Contiguous k order, same as B frag: lanes 0-15 -> M=lane, K=0..15; lanes 16-31 -> K=16..31.
__device__ __forceinline__ v8u load_a_frag(const unsigned short* src, int ld) {
    int lane = threadIdx.x & 31;
    int half = lane >> 4, row = lane & 15;
    const unsigned short* p = src + row * ld;
    v8u a;
#pragma unroll
    for (int v = 0; v < 4; ++v)
        a[v] = *(const unsigned int*)(p + 16 * half + 2 * v);
#pragma unroll
    for (int v = 0; v < 4; ++v)
        a[v + 4] = *(const unsigned int*)(p + 16 * half + 8 + 2 * v);
    return a;
}

// B-fragment (32x16 bf16, K x N) from N-major src: src[n*ld + k].
// Contiguous k order, same as A frag: lanes 0-15 hold K=0..15, lanes 16-31 hold K=16..31, col = lane&15.
__device__ __forceinline__ v8u load_b_frag_t(const unsigned short* src, int ld) {
    int lane = threadIdx.x & 31;
    int half = lane >> 4, col = lane & 15;
    const unsigned short* p = src + col * ld + 16 * half;
    v8u b;
#pragma unroll
    for (int v = 0; v < 8; ++v)
        b[v] = *(const unsigned int*)(p + 2 * v);
    return b;
}

// ---------------------------------------------------------------------------
// RMSNorm + cast to bf16.  One block per token, 256 threads, D = 1024.
// ---------------------------------------------------------------------------
__global__ __launch_bounds__(256)
void rmsnorm_kernel(const float* __restrict__ x, const float* __restrict__ gamma,
                    unsigned short* __restrict__ hBf) {
    __shared__ float red[256];
    long t = blockIdx.x;
    const float* xr = x + t * 1024;
    float vals[4];
    float ss = 0.f;
#pragma unroll
    for (int i = 0; i < 4; ++i) {
        float v = xr[threadIdx.x + i * 256];
        vals[i] = v;
        ss += v * v;
    }
    red[threadIdx.x] = ss;
    __syncthreads();
    for (int s = 128; s > 0; s >>= 1) {
        if ((int)threadIdx.x < s) red[threadIdx.x] += red[threadIdx.x + s];
        __syncthreads();
    }
    float inv = rsqrtf(red[0] * (1.0f / 1024.0f) + 1e-6f);
#pragma unroll
    for (int i = 0; i < 4; ++i) {
        int d = threadIdx.x + i * 256;
        hBf[t * 1024 + d] = f2bf(vals[i] * inv * gamma[d]);
    }
}

__global__ void convert_f32_bf16(const float* __restrict__ in,
                                 unsigned short* __restrict__ out, long n) {
    long i = (long)blockIdx.x * blockDim.x + threadIdx.x;
    if (i < n) out[i] = f2bf(in[i]);
}

// ---------------------------------------------------------------------------
// bf16 GEMM: C[MxN] = A[MxK] * B[KxN], M,N % 128 == 0, K % 64 == 0.
// 256 threads = 8 waves; block tile 128x128 x K-tile 64; wave tile 32x64.
// Double-buffered LDS; A tile staged with async global->LDS b128 copies,
// B tile transposed via VGPR gather. One barrier per K-tile.
// ---------------------------------------------------------------------------
template <bool OUT_F32>
__global__ __launch_bounds__(256)
void gemm_bf16_kernel(const unsigned short* __restrict__ A,
                      const unsigned short* __restrict__ B,
                      void* __restrict__ Cout, int M, int N, int K) {
    __shared__ __align__(16) unsigned short ldsA[2][128 * 64];   // [m][k]
    __shared__ __align__(16) unsigned short ldsBt[2][128 * 64];  // [n][k]

    int tid = threadIdx.x;
    int lane = tid & 31, wave = tid >> 5;
    int wm = (wave & 3) * 32;   // wave row offset in 128-tile
    int wn = (wave >> 2) * 64;  // wave col offset
    long rowBase = (long)blockIdx.y * 128;
    long colBase = (long)blockIdx.x * 128;

    v8f acc[2][4];
#pragma unroll
    for (int mi = 0; mi < 2; ++mi)
#pragma unroll
        for (int ni = 0; ni < 4; ++ni)
            acc[mi][ni] = (v8f){0, 0, 0, 0, 0, 0, 0, 0};

    int nK = K >> 6;
    // ---- stage k-tile 0 into buffer 0 ----
    {
        const unsigned short* ga = A + rowBase * K;
#pragma unroll
        for (int i = 0; i < 4; ++i) {
            int idx = tid + i * 256;          // 0..1023, 16B chunks
            int r = idx >> 3, c = idx & 7;
            async_copy_b128(lds_addr(&ldsA[0][r * 64 + c * 8]),
                            ga + (long)r * K + c * 8);
        }
        const unsigned short* gb = B + colBase;
#pragma unroll
        for (int i = 0; i < 16; ++i) {
            int idx = tid + i * 256;          // 0..4095
            int kk = idx >> 6, u = idx & 63;
            unsigned int d = *(const unsigned int*)(gb + (long)kk * N + 2 * u);
            ldsBt[0][(2 * u + 0) * 64 + kk] = (unsigned short)(d & 0xFFFF);
            ldsBt[0][(2 * u + 1) * 64 + kk] = (unsigned short)(d >> 16);
        }
    }
    wait_async0();
    __syncthreads();

    for (int kt = 0; kt < nK; ++kt) {
        int cur = kt & 1;
        // ---- stage next k-tile into the other buffer ----
        if (kt + 1 < nK) {
            int k0 = (kt + 1) << 6;
            const unsigned short* ga = A + rowBase * K + k0;
#pragma unroll
            for (int i = 0; i < 4; ++i) {
                int idx = tid + i * 256;
                int r = idx >> 3, c = idx & 7;
                async_copy_b128(lds_addr(&ldsA[cur ^ 1][r * 64 + c * 8]),
                                ga + (long)r * K + c * 8);
            }
            const unsigned short* gb = B + (long)k0 * N + colBase;
#pragma unroll
            for (int i = 0; i < 16; ++i) {
                int idx = tid + i * 256;
                int kk = idx >> 6, u = idx & 63;
                unsigned int d = *(const unsigned int*)(gb + (long)kk * N + 2 * u);
                ldsBt[cur ^ 1][(2 * u + 0) * 64 + kk] = (unsigned short)(d & 0xFFFF);
                ldsBt[cur ^ 1][(2 * u + 1) * 64 + kk] = (unsigned short)(d >> 16);
            }
        }
        // ---- 16 WMMAs on current buffer ----
#pragma unroll
        for (int ks = 0; ks < 2; ++ks) {
            v8u af[2], bfr[4];
#pragma unroll
            for (int mi = 0; mi < 2; ++mi)
                af[mi] = load_a_frag(&ldsA[cur][(wm + 16 * mi) * 64 + 32 * ks], 64);
#pragma unroll
            for (int ni = 0; ni < 4; ++ni)
                bfr[ni] = load_b_frag_t(&ldsBt[cur][(wn + 16 * ni) * 64 + 32 * ks], 64);
#pragma unroll
            for (int mi = 0; mi < 2; ++mi)
#pragma unroll
                for (int ni = 0; ni < 4; ++ni)
                    acc[mi][ni] = wmma_bf16(af[mi], bfr[ni], acc[mi][ni]);
        }
        wait_async0();
        __syncthreads();
    }

    int halfl = lane >> 4, nl = lane & 15;
#pragma unroll
    for (int mi = 0; mi < 2; ++mi)
#pragma unroll
        for (int ni = 0; ni < 4; ++ni)
#pragma unroll
            for (int r = 0; r < 8; ++r) {
                long row = rowBase + wm + 16 * mi + r + 8 * halfl;
                long col = colBase + wn + 16 * ni + nl;
                float val = acc[mi][ni][r];
                if (OUT_F32)
                    ((float*)Cout)[row * N + col] = val;
                else
                    ((unsigned short*)Cout)[row * N + col] = f2bf(val);
            }
}

// ---------------------------------------------------------------------------
// Windowed causal attention with persistent-memory K/V (flash-style).
// Computes S^T = K*Q^T so each lane owns ONE query column: softmax reductions
// are 16 in-lane ops + one shfl_xor(16).  K tile staged via async-to-LDS.
// qkv: bf16 [8192][3072]. Out: bf16 [8192][1024].
// Grid: 2048 blocks = (b,w,h) * 8 query-chunks of 64 rows; 128 threads = 4 waves.
// ---------------------------------------------------------------------------
#define PSTRIDE 34   // padded row stride (ushorts) of P bounce buffer

__global__ __launch_bounds__(128)
void attn_kernel(const unsigned short* __restrict__ qkv,
                 const float* __restrict__ pmk, const float* __restrict__ pmv,
                 unsigned short* __restrict__ attOut) {
    __shared__ __align__(16) unsigned short ldsK[32 * 64];       // [j][d]
    __shared__ __align__(16) unsigned short ldsVt[64 * 32];      // [d][j]
    __shared__ __align__(16) unsigned short pLds[4 * 16 * PSTRIDE];

    const float scale = 0.125f;  // DIM_HEAD^-0.5

    int tid = threadIdx.x, lane = tid & 31, wave = tid >> 5;
    int halfl = lane >> 4, rl = lane & 15;

    int qb  = blockIdx.x & 7;
    int bwh = blockIdx.x >> 3;
    int h = bwh & 15;
    int w = (bwh >> 4) & 7;
    int b = bwh >> 7;
    long tokBase = ((long)b * 8 + w) * 512;
    int qbase = qb * 64 + wave * 16;
    int qi = qbase + rl;              // this lane's query (replicated across halves)

    // Q^T B-fragments (n = query col, k = d), straight from global bf16.
    const unsigned short* qrow = qkv + (tokBase + qbase) * (long)3072 + h * 64;
    v8u bq0 = load_b_frag_t(qrow, 3072);        // d 0..31
    v8u bq1 = load_b_frag_t(qrow + 32, 3072);   // d 32..63

    float m = -1e30f, l = 0.f;        // per-lane scalars (one query per lane)
    v8f o[4];
#pragma unroll
    for (int dt = 0; dt < 4; ++dt) o[dt] = (v8f){0, 0, 0, 0, 0, 0, 0, 0};

    unsigned short* pW = pLds + wave * 16 * PSTRIDE;
    int nTiles = 2 * qb + 2;  // causal: key tiles of 32 up to qbase0+63

    for (int it = -1; it < nTiles; ++it) {
        __syncthreads();  // previous iter done reading ldsK/ldsVt
        if (it < 0) {
            // Persistent-memory tile: 16 real K/V rows + 16 zero-padded rows.
            const float* kp = pmk + h * 16 * 64;
            const float* vp = pmv + h * 16 * 64;
#pragma unroll
            for (int i = 0; i < 8; ++i) {
                int idx = tid + i * 128;       // 0..1023
                int p = idx >> 6, d = idx & 63;
                ldsK[p * 64 + d] = f2bf(kp[idx]);
                ldsVt[d * 32 + p] = f2bf(vp[idx]);
                ldsK[(p + 16) * 64 + d] = 0;
                ldsVt[d * 32 + (p + 16)] = 0;
            }
        } else {
            int jr0 = it * 32;
            const unsigned short* kg =
                qkv + (tokBase + jr0) * (long)3072 + 1024 + h * 64;
            const unsigned short* vg = kg + 1024;
            // K tile: straight row copy -> async global->LDS, 16B per lane.
#pragma unroll
            for (int i = 0; i < 2; ++i) {
                int idx = tid + i * 128;       // 0..255, 16B chunks
                int j = idx >> 3, c = idx & 7;
                async_copy_b128(lds_addr(&ldsK[j * 64 + c * 8]),
                                kg + (long)j * 3072 + c * 8);
            }
            // V tile: transpose gather into [d][j].
#pragma unroll
            for (int i = 0; i < 8; ++i) {
                int idx = tid + i * 128;       // 0..1023; 32 uints per 64-elem row
                int j = idx >> 5, du = idx & 31;
                unsigned int vd = *(const unsigned int*)(vg + (long)j * 3072 + 2 * du);
                ldsVt[(2 * du + 0) * 32 + j] = (unsigned short)(vd & 0xFFFF);
                ldsVt[(2 * du + 1) * 32 + j] = (unsigned short)(vd >> 16);
            }
        }
        // Prefetch next tile's K/V rows.
        if (it + 1 < nTiles) {
            int jn = (it + 1) * 32;
            int rsel = tid & 31;
            const unsigned short* nb =
                qkv + (tokBase + jn + rsel) * (long)3072 + h * 64;
            if (tid < 32)       __builtin_prefetch(nb + 1024, 0, 1);
            else if (tid < 64)  __builtin_prefetch(nb + 2048, 0, 1);
        }
        wait_async0();
        __syncthreads();

        // S^T = K Q^T : two j-tiles of 16, K-dim d=64 (two wmma each).
        float sv[2][8];
#pragma unroll
        for (int t = 0; t < 2; ++t) {
            v8u a0 = load_a_frag(ldsK + 16 * t * 64, 64);        // d 0..31
            v8u a1 = load_a_frag(ldsK + 16 * t * 64 + 32, 64);   // d 32..63
            v8f c = wmma_bf16(a0, bq0, (v8f){0, 0, 0, 0, 0, 0, 0, 0});
            c = wmma_bf16(a1, bq1, c);
#pragma unroll
            for (int r = 0; r < 8; ++r) {
                int jloc = 16 * t + r + 8 * halfl;       // key index in tile pair
                float vv = c[r] * scale;
                bool vis = (it < 0) ? (t == 0)           // memory keys always visible
                                    : (qi >= it * 32 + jloc);
                sv[t][r] = vis ? vv : -1e30f;
            }
        }

        // Online softmax: per-lane (per-query) scalars; one shfl_xor each.
        float mx = -1e30f;
#pragma unroll
        for (int t = 0; t < 2; ++t)
#pragma unroll
            for (int r = 0; r < 8; ++r) mx = fmaxf(mx, sv[t][r]);
        mx = fmaxf(mx, __shfl_xor(mx, 16, 32));
        float mnew = fmaxf(m, mx);
        float a = __expf(m - mnew);
        float rs = 0.f;
#pragma unroll
        for (int t = 0; t < 2; ++t)
#pragma unroll
            for (int r = 0; r < 8; ++r) {
                float p = __expf(sv[t][r] - mnew);
                rs += p;
                int jloc = 16 * t + r + 8 * halfl;
                pW[rl * PSTRIDE + jloc] = f2bf(p);       // P[qi_local][j]
            }
        rs += __shfl_xor(rs, 16, 32);
        l = l * a + rs;
        m = mnew;

        // Broadcast alpha into O's row layout (row r+8*half of this half).
        float aRow[8];
#pragma unroll
        for (int r = 0; r < 8; ++r) aRow[r] = __shfl(a, r + 8 * halfl, 16);

        __syncthreads();  // pLds cross-lane visibility

        // O = diag(alpha)*O + P V
        v8u ap = load_a_frag(pW, PSTRIDE);
#pragma unroll
        for (int dt = 0; dt < 4; ++dt) {
            v8u bv = load_b_frag_t(ldsVt + 16 * dt * 32, 32);
            v8f oo = o[dt];
#pragma unroll
            for (int r = 0; r < 8; ++r) oo[r] *= aRow[r];
            o[dt] = wmma_bf16(ap, bv, oo);
        }
    }

    // Normalize and store bf16 [token][h*64 + d].
    float lRow[8];
#pragma unroll
    for (int r = 0; r < 8; ++r) lRow[r] = __shfl(l, r + 8 * halfl, 16);
#pragma unroll
    for (int dt = 0; dt < 4; ++dt)
#pragma unroll
        for (int r = 0; r < 8; ++r) {
            long tok = tokBase + qbase + r + 8 * halfl;
            int col = h * 64 + 16 * dt + rl;
            attOut[tok * 1024 + col] = f2bf(o[dt][r] / lRow[r]);
        }
}

// ---------------------------------------------------------------------------
extern "C" void kernel_launch(void* const* d_in, const int* in_sizes, int n_in,
                              void* d_out, int out_size, void* d_ws, size_t ws_size,
                              hipStream_t stream) {
    (void)in_sizes; (void)n_in; (void)out_size; (void)ws_size;
    const float* x     = (const float*)d_in[0];
    const float* gamma = (const float*)d_in[1];
    const float* w_qkv = (const float*)d_in[2];
    const float* w_out = (const float*)d_in[3];
    const float* pm_k  = (const float*)d_in[4];
    const float* pm_v  = (const float*)d_in[5];

    char* ws = (char*)d_ws;
    size_t off = 0;
    unsigned short* hBf = (unsigned short*)(ws + off);    off += 8192L * 1024 * 2;   // 16 MB
    unsigned short* wqkvBf = (unsigned short*)(ws + off); off += 1024L * 3072 * 2;   // 6 MB
    unsigned short* woutBf = (unsigned short*)(ws + off); off += 1024L * 1024 * 2;   // 2 MB
    unsigned short* qkvBf = (unsigned short*)(ws + off);  off += 8192L * 3072 * 2;   // 48 MB
    unsigned short* attBf = (unsigned short*)(ws + off);  off += 8192L * 1024 * 2;   // 16 MB

    rmsnorm_kernel<<<8192, 256, 0, stream>>>(x, gamma, hBf);
    convert_f32_bf16<<<(3145728 + 255) / 256, 256, 0, stream>>>(w_qkv, wqkvBf, 3145728L);
    convert_f32_bf16<<<(1048576 + 255) / 256, 256, 0, stream>>>(w_out, woutBf, 1048576L);

    gemm_bf16_kernel<false><<<dim3(3072 / 128, 8192 / 128), 256, 0, stream>>>(
        hBf, wqkvBf, (void*)qkvBf, 8192, 3072, 1024);

    attn_kernel<<<2048, 128, 0, stream>>>(qkvBf, pm_k, pm_v, attBf);

    gemm_bf16_kernel<true><<<dim3(1024 / 128, 8192 / 128), 256, 0, stream>>>(
        attBf, woutBf, d_out, 8192, 1024, 1024);
}
